// MultiLevelGraphLayerFull_85143431675974
// MI455X (gfx1250) — matrix operand order repaired and checked by reading, not compiled, not run on hardware
//
#include <hip/hip_runtime.h>

#define DD 128
#define NCELLS 512
#define NGENES 256
#define NHEADS 4
#define HDIM 32
#define ESP 8192
#define EGRN 2048

typedef __attribute__((ext_vector_type(16))) _Float16 v16h;
typedef __attribute__((ext_vector_type(8)))  float    v8f;
typedef __attribute__((__vector_size__(16))) int      v4i;

union V16H { v16h v; _Float16 h[16]; unsigned u[8]; };
union V8F  { v8f  v; float f[8]; };
union H2   { unsigned u; _Float16 h[2]; };

#if defined(__has_builtin)
#if __has_builtin(__builtin_amdgcn_global_load_async_to_lds_b128)
#define USE_ASYNC_LDS 1
#endif
#endif

// global generic pointer -> addrspace(1) v4i* (via integer, C-style addrspace casts are rejected)
#define AS1_V4I(p) ((__attribute__((address_space(1))) v4i*)(unsigned long long)(const void*)(p))
// LDS generic pointer -> addrspace(3) v4i* (low 32 bits of generic LDS address == LDS offset)
#define AS3_V4I(p) ((__attribute__((address_space(3))) v4i*)(unsigned)(unsigned long long)(const void*)(p))

__device__ __forceinline__ void wait_async0() {
#if defined(__has_builtin)
#if __has_builtin(__builtin_amdgcn_s_wait_asynccnt)
  __builtin_amdgcn_s_wait_asynccnt(0);
#else
  asm volatile("s_wait_asynccnt 0x0" ::: "memory");
#endif
#else
  asm volatile("s_wait_asynccnt 0x0" ::: "memory");
#endif
}

// -------------------- WMMA helpers (CDNA5 gfx1250, wave32) --------------------

__device__ __forceinline__ v8f wmma16(v16h a, v16h b, v8f c) {
  // D = A(16x32 f16) x B(32x16 f16) + C(16x16 f32)
  return __builtin_amdgcn_wmma_f32_16x16x32_f16(false, a, false, b, (short)0, c, false, false);
}

// A-fragment (16x32 f16) from a row-major f16 tile in LDS.
// Lane L holds row M = row0 + (L&15); VGPR v holds K pair:
//   k = kc*32 + (L>>4)*16 + v*2  (elements k, k+1 packed)
__device__ __forceinline__ v16h load_a_frag(const _Float16* X, int ldk, int row0, int kc, int lane) {
  V16H a;
  const _Float16* p = X + (row0 + (lane & 15)) * ldk + kc * 32 + ((lane >> 4) << 4);
#pragma unroll
  for (int v = 0; v < 8; ++v) {
    int k = v << 1;
    a.u[v] = *(const unsigned*)(p + k);
  }
  return a.v;
}

// B-fragment: pre-swizzled by prep_frag_kernel; 32 bytes contiguous per lane.
__device__ __forceinline__ v16h load_b_frag(const _Float16* frag, int kc, int nt, int numNt, int lane) {
  return *(const v16h*)(frag + ((((kc * numNt + nt) << 5) + lane) << 4));
}

__device__ __forceinline__ float gelu_exact(float x) {
  return 0.5f * x * (1.0f + erff(x * 0.70710678118654752f));
}
__device__ __forceinline__ float sigm(float x) {
  return 1.0f / (1.0f + __expf(-x));
}

// -------------------- weight fragment pre-swizzle --------------------
// B-matrix 32x16 f16 layout used here (same k order as load_a_frag): lane L: N = nt*16 + (L&15);
// VGPR v holds K = kc*32 + (L>>4)*16 + 2v (+1).
// frag index s = ((kc*numNt + nt)*32 + lane)*16 + 2v + which
__global__ void prep_frag_kernel(const float* __restrict__ W, _Float16* __restrict__ frag, int K, int N) {
  int total = K * N;
  int numNt = N >> 4;
  for (int s = blockIdx.x * blockDim.x + threadIdx.x; s < total; s += gridDim.x * blockDim.x) {
    int which = s & 1;
    int v     = (s >> 1) & 7;
    int lane  = (s >> 4) & 31;
    int rest  = s >> 9;
    int nt    = rest % numNt;
    int kc    = rest / numNt;
    int k = kc * 32 + ((lane >> 4) << 4) + (v << 1) + which;
    int n = (nt << 4) + (lane & 15);
    frag[s] = (_Float16)W[k * N + n];
  }
}

// -------------------- CSR of GRN edges by destination gene (deterministic) --------------------
__global__ void csr_kernel(const int* __restrict__ grn, int* __restrict__ off,
                           int* __restrict__ eid, int* __restrict__ srcb) {
  __shared__ int soff[NGENES + 1];
  int g = threadIdx.x;  // 256 threads
  int c = 0;
  for (int e = 0; e < EGRN; ++e)
    if (grn[EGRN + e] == g) c++;
  soff[g + 1] = c;
  __syncthreads();
  if (g == 0) {
    soff[0] = 0;
    for (int i = 0; i < NGENES; ++i) soff[i + 1] += soff[i];
    for (int i = 0; i <= NGENES; ++i) off[i] = soff[i];
  }
  __syncthreads();
  int p = soff[g];
  for (int e = 0; e < EGRN; ++e) {
    if (grn[EGRN + e] == g) {
      eid[p] = e;
      srcb[p] = grn[e];
      p++;
    }
  }
}

// -------------------- GIN aggregation: h = (1+eps)*x + segment_sum --------------------
__global__ void gin_agg_kernel(const float* __restrict__ high_emb, const int* __restrict__ sedge,
                               const float* __restrict__ eps, float* __restrict__ hbuf) {
  int c = blockIdx.x;       // cell
  int d = threadIdx.x;      // dim (128)
  float acc = 0.0f;
  for (int e = 0; e < ESP; ++e) {
    if (sedge[ESP + e] == c) acc += high_emb[sedge[e] * DD + d];
  }
  hbuf[c * DD + d] = (1.0f + eps[0]) * high_emb[c * DD + d] + acc;
}

// -------------------- generic WMMA GEMM: Y = act(X @ W + b) --------------------
// Compile-time K, N, ACT so the WMMA chain fully unrolls and B-fragments hoist.
// block = 128 threads (4 waves), 16 rows per block; waves split N-tiles.
template <int K, int N, int ACT>
__global__ __launch_bounds__(128) void gemm_kernel(const float* __restrict__ X,
                                                   const _Float16* __restrict__ frag,
                                                   const float* __restrict__ bias,
                                                   float* __restrict__ Y) {
  __shared__ _Float16 sA[16 * 256];
  int tid = threadIdx.x, w = tid >> 5, lane = tid & 31;
  int row0 = blockIdx.x * 16;
  constexpr int numNt = N >> 4;
  constexpr int numKc = K >> 5;
  constexpr int ntPerWave = numNt >> 2;  // 4 waves
#pragma unroll
  for (int i = 0; i < (16 * K) / (128 * 4); ++i) {
    int idx = (i * 128 + tid) * 4;
    float4 v = *(const float4*)(X + row0 * K + idx);
    sA[idx + 0] = (_Float16)v.x;
    sA[idx + 1] = (_Float16)v.y;
    sA[idx + 2] = (_Float16)v.z;
    sA[idx + 3] = (_Float16)v.w;
  }
  __syncthreads();
#pragma unroll
  for (int j = 0; j < ntPerWave; ++j) {
    int nt = w * ntPerWave + j;
    v16h bf[numKc];
#pragma unroll
    for (int kc = 0; kc < numKc; ++kc) bf[kc] = load_b_frag(frag, kc, nt, numNt, lane);
    float b = bias[nt * 16 + (lane & 15)];
    V8F c;
#pragma unroll
    for (int r = 0; r < 8; ++r) c.f[r] = b;
#pragma unroll
    for (int kc = 0; kc < numKc; ++kc) {
      v16h a = load_a_frag(sA, K, 0, kc, lane);
      c.v = wmma16(a, bf[kc], c.v);
    }
#pragma unroll
    for (int r = 0; r < 8; ++r) {
      int m = row0 + r + ((lane >> 4) << 3);
      int n = nt * 16 + (lane & 15);
      float x = c.f[r];
      if (ACT == 1) x = gelu_exact(x);
      Y[m * N + n] = x;
    }
  }
}

// -------------------- LayerNorm over rows of 128 (wave per row) --------------------
__global__ void ln128_kernel(const float* __restrict__ X, const float* __restrict__ g,
                             const float* __restrict__ b, float* __restrict__ Y, int rows) {
  int w = threadIdx.x >> 5, lane = threadIdx.x & 31;
  int row = blockIdx.x * 8 + w;
  if (row >= rows) return;
  int d0 = lane * 4;
  float v0 = X[row * DD + d0 + 0], v1 = X[row * DD + d0 + 1];
  float v2 = X[row * DD + d0 + 2], v3 = X[row * DD + d0 + 3];
  float s = v0 + v1 + v2 + v3;
  float s2 = v0 * v0 + v1 * v1 + v2 * v2 + v3 * v3;
#pragma unroll
  for (int m = 16; m >= 1; m >>= 1) {
    s += __shfl_xor(s, m, 32);
    s2 += __shfl_xor(s2, m, 32);
  }
  float mean = s * (1.0f / 128.0f);
  float var = s2 * (1.0f / 128.0f) - mean * mean;
  float inv = rsqrtf(var + 1e-5f);
  Y[row * DD + d0 + 0] = (v0 - mean) * inv * g[d0 + 0] + b[d0 + 0];
  Y[row * DD + d0 + 1] = (v1 - mean) * inv * g[d0 + 1] + b[d0 + 1];
  Y[row * DD + d0 + 2] = (v2 - mean) * inv * g[d0 + 2] + b[d0 + 2];
  Y[row * DD + d0 + 3] = (v3 - mean) * inv * g[d0 + 3] + b[d0 + 3];
}

// -------------------- fused per-cell TransformerConv --------------------
// One block per cell. 8 waves; wave w owns output dims [w*16, w*16+16).
// LDS map (240 KB): sX[0,64K) sQ[64K,128K) sKV[128K,192K) sAlpha[192K,224K) sRed[224K,240K)
// Async f32 staging region [64K,192K) overlaps sQ/sKV (dead during phase 1).
__global__ __launch_bounds__(256) void cell_kernel(
    const float* __restrict__ low_emb, const int* __restrict__ grn,
    const _Float16* __restrict__ fr_wq, const _Float16* __restrict__ fr_wk,
    const _Float16* __restrict__ fr_wv, const _Float16* __restrict__ fr_wsk,
    const float* __restrict__ bq, const float* __restrict__ bk,
    const float* __restrict__ bv, const float* __restrict__ bsk,
    const float* __restrict__ ln_g, const float* __restrict__ ln_b,
    const int* __restrict__ csr_off, const int* __restrict__ csr_eid,
    const int* __restrict__ csr_src,
    float* __restrict__ lowout, float* __restrict__ gene_mean) {
  __shared__ __align__(16) char smem[245760];
  _Float16* sX    = (_Float16*)(smem);
  _Float16* sQ    = (_Float16*)(smem + 65536);
  _Float16* sKV   = (_Float16*)(smem + 131072);
  float*    sAlpha = (float*)(smem + 196608);
  float*    sRed   = (float*)(smem + 229376);

  int c = blockIdx.x;
  int tid = threadIdx.x, w = tid >> 5, lane = tid & 31;
  const float* Xg = low_emb + (size_t)c * (NGENES * DD);

  // ---- phase 1: load cell's gene matrix into LDS as f16 ----
#if defined(USE_ASYNC_LDS)
  {
    float* sStage = (float*)(smem + 65536);  // 128 KB f32 staging (overlaps sQ/sKV)
#pragma unroll
    for (int i = 0; i < (NGENES * DD) / (256 * 4); ++i) {
      int idx = (i * 256 + tid) * 4;
      __builtin_amdgcn_global_load_async_to_lds_b128(AS1_V4I(Xg + idx), AS3_V4I(sStage + idx), 0, 0);
    }
    wait_async0();
    __syncthreads();
#pragma unroll
    for (int i = 0; i < (NGENES * DD) / (256 * 4); ++i) {
      int idx = (i * 256 + tid) * 4;
      float4 v = *(const float4*)(sStage + idx);
      sX[idx + 0] = (_Float16)v.x;
      sX[idx + 1] = (_Float16)v.y;
      sX[idx + 2] = (_Float16)v.z;
      sX[idx + 3] = (_Float16)v.w;
    }
  }
#else
#pragma unroll
  for (int i = 0; i < (NGENES * DD) / (256 * 4); ++i) {
    int idx = (i * 256 + tid) * 4;
    float4 v = *(const float4*)(Xg + idx);
    sX[idx + 0] = (_Float16)v.x;
    sX[idx + 1] = (_Float16)v.y;
    sX[idx + 2] = (_Float16)v.z;
    sX[idx + 3] = (_Float16)v.w;
  }
#endif
  __syncthreads();

  int nt = w;                         // this wave's N-tile
  int nn = nt * 16 + (lane & 15);     // this lane's output dim

  // ---- phase 2: Q and K via WMMA (wave computes its 16-wide slice for all genes) ----
  {
    v16h bqf[4], bkf[4];
#pragma unroll
    for (int kc = 0; kc < 4; ++kc) {
      bqf[kc] = load_b_frag(fr_wq, kc, nt, 8, lane);
      bkf[kc] = load_b_frag(fr_wk, kc, nt, 8, lane);
    }
    float biasq = bq[nn], biask = bk[nn];
    for (int mt = 0; mt < 16; ++mt) {
      V8F cq, ck;
#pragma unroll
      for (int r = 0; r < 8; ++r) { cq.f[r] = biasq; ck.f[r] = biask; }
#pragma unroll
      for (int kc = 0; kc < 4; ++kc) {
        v16h a = load_a_frag(sX, DD, mt * 16, kc, lane);
        cq.v = wmma16(a, bqf[kc], cq.v);
        ck.v = wmma16(a, bkf[kc], ck.v);
      }
#pragma unroll
      for (int r = 0; r < 8; ++r) {
        int row = mt * 16 + r + ((lane >> 4) << 3);
        sQ[row * DD + nn] = (_Float16)cq.f[r];
        sKV[row * DD + nn] = (_Float16)ck.f[r];
      }
    }
  }
  __syncthreads();

  // ---- phase 3a: per-edge per-head logits (packed f16x2 LDS reads) ----
  for (int t = tid; t < EGRN * NHEADS; t += 256) {
    int e = t >> 2, h = t & 3;
    int src = grn[e], dst = grn[EGRN + e];
    const unsigned* qp = (const unsigned*)(sQ + dst * DD + h * HDIM);
    const unsigned* kp = (const unsigned*)(sKV + src * DD + h * HDIM);
    float d = 0.0f;
#pragma unroll
    for (int j = 0; j < HDIM / 2; ++j) {
      H2 qa, ka;
      qa.u = qp[j];
      ka.u = kp[j];
      d += (float)qa.h[0] * (float)ka.h[0] + (float)qa.h[1] * (float)ka.h[1];
    }
    sAlpha[e * 4 + h] = d * 0.17677669529663687f;  // 1/sqrt(32)
  }
  __syncthreads();

  // ---- phase 3b: softmax per (dst gene, head), deterministic CSR order ----
  for (int t = tid; t < NGENES * NHEADS; t += 256) {
    int g = t >> 2, h = t & 3;
    int s0 = csr_off[g], s1 = csr_off[g + 1];
    if (s0 == s1) continue;
    float m = -3.402823466e38f;
    for (int i = s0; i < s1; ++i) m = fmaxf(m, sAlpha[csr_eid[i] * 4 + h]);
    float den = 0.0f;
    for (int i = s0; i < s1; ++i) {
      float ex = __expf(sAlpha[csr_eid[i] * 4 + h] - m);
      sAlpha[csr_eid[i] * 4 + h] = ex;
      den += ex;
    }
    float invd = 1.0f / den;
    for (int i = s0; i < s1; ++i) sAlpha[csr_eid[i] * 4 + h] *= invd;
  }
  __syncthreads();

  // ---- phase 4: V via WMMA into sKV (K is dead) ----
  {
    v16h bvf[4];
#pragma unroll
    for (int kc = 0; kc < 4; ++kc) bvf[kc] = load_b_frag(fr_wv, kc, nt, 8, lane);
    float biasv = bv[nn];
    for (int mt = 0; mt < 16; ++mt) {
      V8F cv;
#pragma unroll
      for (int r = 0; r < 8; ++r) cv.f[r] = biasv;
#pragma unroll
      for (int kc = 0; kc < 4; ++kc) {
        v16h a = load_a_frag(sX, DD, mt * 16, kc, lane);
        cv.v = wmma16(a, bvf[kc], cv.v);
      }
#pragma unroll
      for (int r = 0; r < 8; ++r) {
        int row = mt * 16 + r + ((lane >> 4) << 3);
        sKV[row * DD + nn] = (_Float16)cv.f[r];
      }
    }
  }
  __syncthreads();

  // ---- phase 5: out = skip-GEMM (WMMA) + attention aggregation, in C-layout regs ----
  int hh = nt >> 1;  // head for this wave's dim slice (16-wide slice within one 32-wide head)
  float acc[16][8];
  {
    v16h bsf[4];
#pragma unroll
    for (int kc = 0; kc < 4; ++kc) bsf[kc] = load_b_frag(fr_wsk, kc, nt, 8, lane);
    float biass = bsk[nn];
#pragma unroll
    for (int mt = 0; mt < 16; ++mt) {
      V8F cs;
#pragma unroll
      for (int r = 0; r < 8; ++r) cs.f[r] = biass;
#pragma unroll
      for (int kc = 0; kc < 4; ++kc) {
        v16h a = load_a_frag(sX, DD, mt * 16, kc, lane);
        cs.v = wmma16(a, bsf[kc], cs.v);
      }
#pragma unroll
      for (int r = 0; r < 8; ++r) acc[mt][r] = cs.f[r];
    }
  }
#pragma unroll
  for (int mt = 0; mt < 16; ++mt) {
#pragma unroll
    for (int r = 0; r < 8; ++r) {
      int g = mt * 16 + r + ((lane >> 4) << 3);
      int s0 = csr_off[g], s1 = csr_off[g + 1];
      float a = acc[mt][r];
      for (int i = s0; i < s1; ++i)
        a += sAlpha[csr_eid[i] * 4 + hh] * (float)sKV[csr_src[i] * DD + nn];
      acc[mt][r] = a;
    }
  }
  // LN partial sums (per gene, this wave's 16 dims spread over a 16-lane group)
#pragma unroll
  for (int mt = 0; mt < 16; ++mt) {
#pragma unroll
    for (int r = 0; r < 8; ++r) {
      float v = acc[mt][r];
      float s = v, s2 = v * v;
#pragma unroll
      for (int m = 8; m >= 1; m >>= 1) {
        s += __shfl_xor(s, m, 32);
        s2 += __shfl_xor(s2, m, 32);
      }
      if ((lane & 15) == 0) {
        int g = mt * 16 + r + ((lane >> 4) << 3);
        sRed[g * 16 + w * 2 + 0] = s;
        sRed[g * 16 + w * 2 + 1] = s2;
      }
    }
  }
  __syncthreads();

  // ---- phase 5b: LayerNorm, write low_out, accumulate gene mean ----
  float lg = ln_g[nn], lb = ln_b[nn];
  float gm = 0.0f;
  float* outc = lowout + (size_t)c * (NGENES * DD);
#pragma unroll
  for (int mt = 0; mt < 16; ++mt) {
#pragma unroll
    for (int r = 0; r < 8; ++r) {
      int g = mt * 16 + r + ((lane >> 4) << 3);
      float S = 0.0f, S2 = 0.0f;
#pragma unroll
      for (int w2 = 0; w2 < 8; ++w2) {
        S += sRed[g * 16 + w2 * 2 + 0];
        S2 += sRed[g * 16 + w2 * 2 + 1];
      }
      float mean = S * (1.0f / 128.0f);
      float var = S2 * (1.0f / 128.0f) - mean * mean;
      float inv = rsqrtf(var + 1e-5f);
      float vn = (acc[mt][r] - mean) * inv * lg + lb;
      outc[g * DD + nn] = vn;
      gm += vn;
    }
  }
  gm += __shfl_xor(gm, 16, 32);
  if (lane < 16) gene_mean[c * DD + nn] = gm * (1.0f / 256.0f);
}

// -------------------- cross gating + high_new --------------------
__global__ void gate_kernel(const float* __restrict__ q1, const float* __restrict__ k1,
                            const float* __restrict__ v1, const float* __restrict__ q2,
                            const float* __restrict__ k2, const float* __restrict__ v2,
                            const float* __restrict__ high_emb, const float* __restrict__ high_out,
                            const float* __restrict__ nh_g, const float* __restrict__ nh_b,
                            float* __restrict__ high_new, float* __restrict__ low_cross) {
  int w = threadIdx.x >> 5, lane = threadIdx.x & 31;
  int row = blockIdx.x * 8 + w;
  if (row >= NCELLS) return;
  int base = row * DD, d0 = lane * 4;
  float d1 = 0.0f, d2 = 0.0f;
#pragma unroll
  for (int j = 0; j < 4; ++j) {
    d1 += q1[base + d0 + j] * k1[base + d0 + j];
    d2 += q2[base + d0 + j] * k2[base + d0 + j];
  }
#pragma unroll
  for (int m = 16; m >= 1; m >>= 1) {
    d1 += __shfl_xor(d1, m, 32);
    d2 += __shfl_xor(d2, m, 32);
  }
  float g1 = sigm(d1 * 0.08838834764831845f);  // D^-0.5
  float g2 = sigm(d2 * 0.08838834764831845f);
  float t[4];
  float s = 0.0f, s2 = 0.0f;
#pragma unroll
  for (int j = 0; j < 4; ++j) {
    low_cross[base + d0 + j] = g2 * v2[base + d0 + j];
    t[j] = high_emb[base + d0 + j] + high_out[base + d0 + j] + g1 * v1[base + d0 + j];
    s += t[j];
    s2 += t[j] * t[j];
  }
#pragma unroll
  for (int m = 16; m >= 1; m >>= 1) {
    s += __shfl_xor(s, m, 32);
    s2 += __shfl_xor(s2, m, 32);
  }
  float mean = s * (1.0f / 128.0f);
  float var = s2 * (1.0f / 128.0f) - mean * mean;
  float inv = rsqrtf(var + 1e-5f);
#pragma unroll
  for (int j = 0; j < 4; ++j)
    high_new[base + d0 + j] = (t[j] - mean) * inv * nh_g[d0 + j] + nh_b[d0 + j];
}

// -------------------- finalize low_new (in place over d_out low region) --------------------
__global__ void final_low_kernel(const float* __restrict__ low_emb, const float* __restrict__ low_cross,
                                 const float* __restrict__ nl_g, const float* __restrict__ nl_b,
                                 float* __restrict__ lowbuf) {
  int w = threadIdx.x >> 5, lane = threadIdx.x & 31;
  long long R = (long long)blockIdx.x * 8 + w;  // cell*256 + gene
  int c = (int)(R >> 8);
  long long base = R * DD;
  int d0 = lane * 4;
  float t[4];
  float s = 0.0f, s2 = 0.0f;
#pragma unroll
  for (int j = 0; j < 4; ++j) {
    t[j] = low_emb[base + d0 + j] + lowbuf[base + d0 + j] + low_cross[c * DD + d0 + j];
    s += t[j];
    s2 += t[j] * t[j];
  }
#pragma unroll
  for (int m = 16; m >= 1; m >>= 1) {
    s += __shfl_xor(s, m, 32);
    s2 += __shfl_xor(s2, m, 32);
  }
  float mean = s * (1.0f / 128.0f);
  float var = s2 * (1.0f / 128.0f) - mean * mean;
  float inv = rsqrtf(var + 1e-5f);
#pragma unroll
  for (int j = 0; j < 4; ++j)
    lowbuf[base + d0 + j] = (t[j] - mean) * inv * nl_g[d0 + j] + nl_b[d0 + j];
}

// -------------------- host orchestration --------------------
extern "C" void kernel_launch(void* const* d_in, const int* in_sizes, int n_in,
                              void* d_out, int out_size, void* d_ws, size_t ws_size,
                              hipStream_t stream) {
  (void)in_sizes; (void)n_in; (void)out_size; (void)ws_size;
  const float* high_emb = (const float*)d_in[0];
  const float* low_emb  = (const float*)d_in[1];
  const int*   sp_edge  = (const int*)d_in[2];
  const int*   grn_edge = (const int*)d_in[3];
  const float* gin_w1 = (const float*)d_in[4];
  const float* gin_b1 = (const float*)d_in[5];
  const float* gin_w2 = (const float*)d_in[6];
  const float* gin_b2 = (const float*)d_in[7];
  const float* gin_eps = (const float*)d_in[8];
  const float* gin_ln_g = (const float*)d_in[9];
  const float* gin_ln_b = (const float*)d_in[10];
  const float* tc_wq = (const float*)d_in[11];
  const float* tc_bq = (const float*)d_in[12];
  const float* tc_wk = (const float*)d_in[13];
  const float* tc_bk = (const float*)d_in[14];
  const float* tc_wv = (const float*)d_in[15];
  const float* tc_bv = (const float*)d_in[16];
  const float* tc_wskip = (const float*)d_in[17];
  const float* tc_bskip = (const float*)d_in[18];
  const float* tc_ln_g = (const float*)d_in[19];
  const float* tc_ln_b = (const float*)d_in[20];
  const float* cross_w = (const float*)d_in[21];
  const float* cross_b = (const float*)d_in[22];
  const float* agg_w = (const float*)d_in[23];
  const float* agg_b = (const float*)d_in[24];
  const float* nh_g = (const float*)d_in[25];
  const float* nh_b = (const float*)d_in[26];
  const float* nl_g = (const float*)d_in[27];
  const float* nl_b = (const float*)d_in[28];

  char* wsb = (char*)d_ws;
  size_t off = 0;
  auto take = [&](size_t bytes) -> void* {
    void* p = wsb + off;
    off = (off + bytes + 255) & ~(size_t)255;
    return p;
  };
  int* csr_off = (int*)take(257 * 4);
  int* csr_eid = (int*)take(EGRN * 4);
  int* csr_src = (int*)take(EGRN * 4);
  _Float16* fr_wq  = (_Float16*)take(128 * 128 * 2);
  _Float16* fr_wk  = (_Float16*)take(128 * 128 * 2);
  _Float16* fr_wv  = (_Float16*)take(128 * 128 * 2);
  _Float16* fr_wsk = (_Float16*)take(128 * 128 * 2);
  _Float16* fr_w1  = (_Float16*)take(128 * 256 * 2);
  _Float16* fr_w2  = (_Float16*)take(256 * 128 * 2);
  _Float16* fr_agg = (_Float16*)take(128 * 128 * 2);
  _Float16* fr_cross = (_Float16*)take(6 * 128 * 128 * 2);
  float* hbuf      = (float*)take(512 * 128 * 4);
  float* tmp1      = (float*)take(512 * 256 * 4);
  float* tmp2      = (float*)take(512 * 128 * 4);
  float* high_out  = (float*)take(512 * 128 * 4);
  float* gene_mean = (float*)take(512 * 128 * 4);
  float* gene_agg  = (float*)take(512 * 128 * 4);
  float* crbuf     = (float*)take(6 * 512 * 128 * 4);
  float* q1 = crbuf + 0 * 65536, *k1 = crbuf + 1 * 65536, *v1 = crbuf + 2 * 65536;
  float* q2 = crbuf + 3 * 65536, *k2 = crbuf + 4 * 65536, *v2 = crbuf + 5 * 65536;
  float* low_cross = (float*)take(512 * 128 * 4);

  float* out_high = (float*)d_out;
  float* out_low  = out_high + NCELLS * DD;

  // weight fragment pre-swizzle
  prep_frag_kernel<<<64, 256, 0, stream>>>(tc_wq, fr_wq, 128, 128);
  prep_frag_kernel<<<64, 256, 0, stream>>>(tc_wk, fr_wk, 128, 128);
  prep_frag_kernel<<<64, 256, 0, stream>>>(tc_wv, fr_wv, 128, 128);
  prep_frag_kernel<<<64, 256, 0, stream>>>(tc_wskip, fr_wsk, 128, 128);
  prep_frag_kernel<<<128, 256, 0, stream>>>(gin_w1, fr_w1, 128, 256);
  prep_frag_kernel<<<128, 256, 0, stream>>>(gin_w2, fr_w2, 256, 128);
  prep_frag_kernel<<<64, 256, 0, stream>>>(agg_w, fr_agg, 128, 128);
  for (int i = 0; i < 6; ++i)
    prep_frag_kernel<<<64, 256, 0, stream>>>(cross_w + i * 16384, fr_cross + i * 16384, 128, 128);

  // GRN CSR (deterministic edge order)
  csr_kernel<<<1, 256, 0, stream>>>(grn_edge, csr_off, csr_eid, csr_src);

  // high level: GIN conv
  gin_agg_kernel<<<NCELLS, 128, 0, stream>>>(high_emb, sp_edge, gin_eps, hbuf);
  gemm_kernel<128, 256, 1><<<32, 128, 0, stream>>>(hbuf, fr_w1, gin_b1, tmp1);
  gemm_kernel<256, 128, 0><<<32, 128, 0, stream>>>(tmp1, fr_w2, gin_b2, tmp2);
  ln128_kernel<<<64, 256, 0, stream>>>(tmp2, gin_ln_g, gin_ln_b, high_out, NCELLS);

  // low level: fused per-cell TransformerConv (writes pre-residual low_out into d_out region)
  cell_kernel<<<NCELLS, 256, 0, stream>>>(low_emb, grn_edge, fr_wq, fr_wk, fr_wv, fr_wsk,
                                          tc_bq, tc_bk, tc_bv, tc_bskip, tc_ln_g, tc_ln_b,
                                          csr_off, csr_eid, csr_src, out_low, gene_mean);

  // gene aggregation + cross projections
  gemm_kernel<128, 128, 1><<<32, 128, 0, stream>>>(gene_mean, fr_agg, agg_b, gene_agg);
  gemm_kernel<128, 128, 0><<<32, 128, 0, stream>>>(high_out, fr_cross + 0 * 16384, cross_b + 0 * 128, q1);
  gemm_kernel<128, 128, 0><<<32, 128, 0, stream>>>(gene_agg, fr_cross + 1 * 16384, cross_b + 1 * 128, k1);
  gemm_kernel<128, 128, 0><<<32, 128, 0, stream>>>(gene_agg, fr_cross + 2 * 16384, cross_b + 2 * 128, v1);
  gemm_kernel<128, 128, 0><<<32, 128, 0, stream>>>(gene_agg, fr_cross + 3 * 16384, cross_b + 3 * 128, q2);
  gemm_kernel<128, 128, 0><<<32, 128, 0, stream>>>(high_out, fr_cross + 4 * 16384, cross_b + 4 * 128, k2);
  gemm_kernel<128, 128, 0><<<32, 128, 0, stream>>>(high_out, fr_cross + 5 * 16384, cross_b + 5 * 128, v2);

  // gating + high_new, then finalize low_new in place
  gate_kernel<<<64, 256, 0, stream>>>(q1, k1, v1, q2, k2, v2, high_emb, high_out,
                                      nh_g, nh_b, out_high, low_cross);
  final_low_kernel<<<NCELLS * NGENES / 8, 256, 0, stream>>>(low_emb, low_cross, nl_g, nl_b, out_low);
}
